// EdgeNetwork_5403068859067
// MI455X (gfx1250) — matrix operand order repaired and checked by reading, not compiled, not run on hardware
//
#include <hip/hip_runtime.h>
#include <hip/hip_bf16.h>

typedef __attribute__((ext_vector_type(16))) _Float16 v16h;
typedef __attribute__((ext_vector_type(8)))  _Float16 h8;
typedef __attribute__((ext_vector_type(2)))  __fp16   fp16x2;   // cvt_pkrtz return type
typedef __attribute__((ext_vector_type(8)))  float    v8f;

#define IN_DIM   261
#define NK1      9            // K chunks, layer 1 (K padded 261 -> 288)
#define NK2      4            // K chunks, layer 2 (K=128)
#define NT       8            // 8 n-tiles of 16 -> 128 outputs
#define HID      128
#define XSTRIDE  296          // padded LDS row stride (halves), 16B-aligned
#define HSTRIDE  136          // padded h row stride (halves), 16B-aligned
#define WAVES    2            // waves per block
#define W1FRAG_ELEMS (NK1*NT*32*16)   // 36864 halves
#define W2FRAG_ELEMS (NK2*NT*32*16)   // 16384 halves

union ABfrag { v16h v; h8 h[2]; };

// Permuted K order for layer-1 input rows (keeps hidden segments 16B aligned):
// K' 0..127   -> dst_hidden  (orig 2..129)
// K' 128..255 -> src_hidden  (orig 132..259)
// K' 256,257  -> dst_feat    (orig 0,1)
// K' 258,259  -> src_feat    (orig 130,131)
// K' 260      -> distance    (orig 260)
// K' 261..287 -> zero pad
__device__ __forceinline__ int korig(int kp) {
    if (kp < 128) return 2 + kp;
    if (kp < 256) return 132 + (kp - 128);
    if (kp == 256) return 0;
    if (kp == 257) return 1;
    if (kp == 258) return 130;
    if (kp == 259) return 131;
    if (kp == 260) return 260;
    return -1;
}

// Pre-swizzle W1 (IN_DIM x 128, f32) and W2 (128 x 128, f32) into f16
// B-fragment layout: frag[kc][nt][lane][i], element i of lane holds
// K = 32*kc + (lane>=16 ? 8 : 0) + (i&7) + (i>=8 ? 16 : 0), N = nt*16 + (lane&15).
__global__ void edgenet_prep_kernel(const float* __restrict__ W1,
                                    const float* __restrict__ W2,
                                    _Float16* __restrict__ w1f,
                                    _Float16* __restrict__ w2f) {
    int t = blockIdx.x * blockDim.x + threadIdx.x;
    if (t < W1FRAG_ELEMS) {
        int i    = t & 15;
        int lane = (t >> 4) & 31;
        int nt   = (t >> 9) & 7;
        int kc   = t >> 12;
        int kp   = 32 * kc + ((lane >= 16) ? 8 : 0) + (i & 7) + ((i >> 3) << 4);
        int n    = nt * 16 + (lane & 15);
        int ko   = korig(kp);
        float v  = (ko >= 0) ? W1[ko * HID + n] : 0.0f;
        w1f[t] = (_Float16)v;
    }
    int t2 = t - W1FRAG_ELEMS;
    if (t2 >= 0 && t2 < W2FRAG_ELEMS) {
        int i    = t2 & 15;
        int lane = (t2 >> 4) & 31;
        int nt   = (t2 >> 9) & 7;
        int kc   = t2 >> 12;
        int k    = 32 * kc + ((lane >= 16) ? 8 : 0) + (i & 7) + ((i >> 3) << 4);   // 0..127
        int n    = nt * 16 + (lane & 15);
        w2f[t2] = (_Float16)W2[k * HID + n];
    }
}

// 128 f32 -> 128 f16 into LDS using packed converts (v_cvt_pk_rtz_f16_f32).
__device__ __forceinline__ void stage_hidden(const float* __restrict__ srcrow,
                                             _Float16* __restrict__ dst) {
    const float4* p = (const float4*)srcrow;
    #pragma unroll
    for (int g = 0; g < 16; ++g) {
        float4 q0 = p[2 * g], q1 = p[2 * g + 1];
        union { h8 v; fp16x2 p2[4]; } u;
        u.p2[0] = __builtin_amdgcn_cvt_pkrtz(q0.x, q0.y);
        u.p2[1] = __builtin_amdgcn_cvt_pkrtz(q0.z, q0.w);
        u.p2[2] = __builtin_amdgcn_cvt_pkrtz(q1.x, q1.y);
        u.p2[3] = __builtin_amdgcn_cvt_pkrtz(q1.z, q1.w);
        *(h8*)(dst + g * 8) = u.v;
    }
}

__global__ void __launch_bounds__(WAVES * 32)
edgenet_main_kernel(const float* __restrict__ node_features,
                    const float* __restrict__ node_hidden,
                    const float* __restrict__ distance,
                    const int*   __restrict__ src_idx,
                    const int*   __restrict__ dst_idx,
                    const float* __restrict__ b1,
                    const float* __restrict__ b2,
                    const _Float16* __restrict__ w1f,
                    const _Float16* __restrict__ w2f,
                    float* __restrict__ out,
                    int E) {
    // Per-wave slice: 32 edge rows of XSTRIDE halves. The h tiles (16 x HSTRIDE
    // per M-tile) reuse the X region after layer 1 consumes it (LDS ops from a
    // wave execute in order, and the aliasing is visible to the compiler).
    __shared__ __align__(16) _Float16 Xs[WAVES * 32 * XSTRIDE];

    const int lane   = threadIdx.x & 31;
    const int waveId = threadIdx.x >> 5;
    const int gwave  = blockIdx.x * WAVES + waveId;
    const int nwaves = gridDim.x * WAVES;

    _Float16* xbase  = Xs + waveId * 32 * XSTRIDE;
    _Float16* xbase2 = xbase + 16 * XSTRIDE;

    const int nsel  = (lane & 15);           // column within 16-wide n tile
    const int kb    = (lane >= 16) ? 8 : 0;  // A fragment K sub-base
    const int mrow  = lane & 15;             // A fragment row
    const int mbase = (lane >= 16) ? 8 : 0;  // C/D row base

    // Loop-invariant bias values (one column per lane, per n-tile).
    float b1r[NT], b2r[NT];
    #pragma unroll
    for (int nt = 0; nt < NT; ++nt) {
        b1r[nt] = b1[nt * 16 + nsel];
        b2r[nt] = b2[nt * 16 + nsel];
    }

    const int stride32 = nwaves * 32;
    for (int base = gwave * 32; base < E; base += stride32) {
        // ---------------- Stage 32 gathered input rows (f32 -> f16) ----------
        {
            const int m  = lane;                       // edge within 32-group
            const int e  = base + m;
            const int ec = (e < E) ? e : (E - 1);      // clamp (tail safety)
            const int nd = dst_idx[ec];
            const int ns = src_idx[ec];
            _Float16* row = xbase + m * XSTRIDE;

            stage_hidden(node_hidden + (size_t)nd * HID, row);        // K' 0..127
            stage_hidden(node_hidden + (size_t)ns * HID, row + 128);  // K' 128..255

            h8 ex;
            ex[0] = (_Float16)node_features[2 * nd];
            ex[1] = (_Float16)node_features[2 * nd + 1];
            ex[2] = (_Float16)node_features[2 * ns];
            ex[3] = (_Float16)node_features[2 * ns + 1];
            ex[4] = (_Float16)distance[ec];
            ex[5] = (_Float16)0.0f; ex[6] = (_Float16)0.0f; ex[7] = (_Float16)0.0f;
            *(h8*)(row + 256) = ex;
            h8 z;
            #pragma unroll
            for (int i = 0; i < 8; ++i) z[i] = (_Float16)0.0f;
            *(h8*)(row + 264) = z;
            *(h8*)(row + 272) = z;
            *(h8*)(row + 280) = z;
        }
        asm volatile("s_wait_dscnt 0x0" ::: "memory");

        // ---------------- Layer 1: two 16-row tiles share every B fragment ---
        v8f acc1[NT], acc2[NT];
        #pragma unroll
        for (int nt = 0; nt < NT; ++nt) {
            float b = b1r[nt];
            #pragma unroll
            for (int r = 0; r < 8; ++r) { acc1[nt][r] = b; acc2[nt][r] = b; }
        }
        {
            const _Float16* xrow1 = xbase  + mrow * XSTRIDE;
            const _Float16* xrow2 = xbase2 + mrow * XSTRIDE;
            #pragma unroll
            for (int kc = 0; kc < NK1; ++kc) {
                ABfrag A1, A2;
                A1.h[0] = *(const h8*)(xrow1 + kc * 32 + kb);
                A1.h[1] = *(const h8*)(xrow1 + kc * 32 + kb + 16);
                A2.h[0] = *(const h8*)(xrow2 + kc * 32 + kb);
                A2.h[1] = *(const h8*)(xrow2 + kc * 32 + kb + 16);
                #pragma unroll
                for (int nt = 0; nt < NT; ++nt) {
                    ABfrag B;
                    const h8* wp = (const h8*)(w1f + ((size_t)(kc * NT + nt) * 32 + lane) * 16);
                    B.h[0] = wp[0];
                    B.h[1] = wp[1];
                    acc1[nt] = __builtin_amdgcn_wmma_f32_16x16x32_f16(
                        false, A1.v, false, B.v, (short)0, acc1[nt], false, false);
                    acc2[nt] = __builtin_amdgcn_wmma_f32_16x16x32_f16(
                        false, A2.v, false, B.v, (short)0, acc2[nt], false, false);
                }
            }
        }
        // relu -> f16 h tiles in LDS (reusing the X region, now dead)
        #pragma unroll
        for (int nt = 0; nt < NT; ++nt) {
            #pragma unroll
            for (int r = 0; r < 8; ++r) {
                float v1 = fmaxf(acc1[nt][r], 0.0f);
                float v2 = fmaxf(acc2[nt][r], 0.0f);
                xbase [(mbase + r) * HSTRIDE + nt * 16 + nsel] = (_Float16)v1;
                xbase2[(mbase + r) * HSTRIDE + nt * 16 + nsel] = (_Float16)v2;
            }
        }
        asm volatile("s_wait_dscnt 0x0" ::: "memory");

        // ---------------- Prefetch next group's node rows into cache --------
        {
            const int eN = base + stride32 + lane;
            if (eN < E) {
                const int ndN = dst_idx[eN];
                const int nsN = src_idx[eN];
                const float* pd = node_hidden + (size_t)ndN * HID;
                const float* ps = node_hidden + (size_t)nsN * HID;
                #pragma unroll
                for (int j = 0; j < 4; ++j) {           // 4 x 128B = 512B row
                    __builtin_prefetch(pd + j * 32, 0, 3);
                    __builtin_prefetch(ps + j * 32, 0, 3);
                }
                __builtin_prefetch(node_features + 2 * ndN, 0, 3);
                __builtin_prefetch(node_features + 2 * nsN, 0, 3);
            }
        }

        // ---------------- Layer 2 ------------------------------------------
        #pragma unroll
        for (int nt = 0; nt < NT; ++nt) {
            float b = b2r[nt];
            #pragma unroll
            for (int r = 0; r < 8; ++r) { acc1[nt][r] = b; acc2[nt][r] = b; }
        }
        {
            const _Float16* hrow1 = xbase  + mrow * HSTRIDE;
            const _Float16* hrow2 = xbase2 + mrow * HSTRIDE;
            #pragma unroll
            for (int kc = 0; kc < NK2; ++kc) {
                ABfrag A1, A2;
                A1.h[0] = *(const h8*)(hrow1 + kc * 32 + kb);
                A1.h[1] = *(const h8*)(hrow1 + kc * 32 + kb + 16);
                A2.h[0] = *(const h8*)(hrow2 + kc * 32 + kb);
                A2.h[1] = *(const h8*)(hrow2 + kc * 32 + kb + 16);
                #pragma unroll
                for (int nt = 0; nt < NT; ++nt) {
                    ABfrag B;
                    const h8* wp = (const h8*)(w2f + ((size_t)(kc * NT + nt) * 32 + lane) * 16);
                    B.h[0] = wp[0];
                    B.h[1] = wp[1];
                    acc1[nt] = __builtin_amdgcn_wmma_f32_16x16x32_f16(
                        false, A1.v, false, B.v, (short)0, acc1[nt], false, false);
                    acc2[nt] = __builtin_amdgcn_wmma_f32_16x16x32_f16(
                        false, A2.v, false, B.v, (short)0, acc2[nt], false, false);
                }
            }
        }
        // relu -> f32 output (each store instruction covers two 64B sectors)
        #pragma unroll
        for (int nt = 0; nt < NT; ++nt) {
            #pragma unroll
            for (int r = 0; r < 8; ++r) {
                int e1 = base + mbase + r;
                int e2 = e1 + 16;
                float v1 = fmaxf(acc1[nt][r], 0.0f);
                float v2 = fmaxf(acc2[nt][r], 0.0f);
                if (e1 < E) out[(size_t)e1 * HID + nt * 16 + nsel] = v1;
                if (e2 < E) out[(size_t)e2 * HID + nt * 16 + nsel] = v2;
            }
        }
        asm volatile("s_wait_dscnt 0x0" ::: "memory");
    }
}

extern "C" void kernel_launch(void* const* d_in, const int* in_sizes, int n_in,
                              void* d_out, int out_size, void* d_ws, size_t ws_size,
                              hipStream_t stream) {
    const float* node_features = (const float*)d_in[0];
    const float* node_hidden   = (const float*)d_in[1];
    const float* distance      = (const float*)d_in[2];
    const int*   src_idx       = (const int*)d_in[3];
    const int*   dst_idx       = (const int*)d_in[4];
    const float* W1            = (const float*)d_in[5];
    const float* b1            = (const float*)d_in[6];
    const float* W2            = (const float*)d_in[7];
    const float* b2            = (const float*)d_in[8];
    float* out = (float*)d_out;

    const int E = in_sizes[2];          // number of edges (800000)

    _Float16* w1f = (_Float16*)d_ws;
    _Float16* w2f = w1f + W1FRAG_ELEMS;

    const int prepTotal = W1FRAG_ELEMS + W2FRAG_ELEMS;
    edgenet_prep_kernel<<<(prepTotal + 255) / 256, 256, 0, stream>>>(W1, W2, w1f, w2f);

    const int nGroups = (E + 31) / 32;              // 32-edge groups
    int blocks = 2048;
    if (blocks * WAVES > nGroups) blocks = (nGroups + WAVES - 1) / WAVES;
    if (blocks < 1) blocks = 1;
    edgenet_main_kernel<<<blocks, WAVES * 32, 0, stream>>>(
        node_features, node_hidden, distance, src_idx, dst_idx,
        b1, b2, w1f, w2f, out, E);
}
